// S2SMLayer_32246614458535
// MI455X (gfx1250) — compile-verified, matrix operand order checked by reading, not run on hardware
//
#include <hip/hip_runtime.h>

#define LSEQ   2048
#define DDIM   256
#define NBATCH 16
#define NFILT  100
#define KSZ    3
#define NEG_HUGE (-3.402823466e38f)

typedef __attribute__((ext_vector_type(16))) __bf16 v16bf;
typedef __attribute__((ext_vector_type(8)))  float  v8f;

union Frag { v16bf v; uint4 q[2]; };

// fp32x2 -> packed bf16x2 in ONE v_perm_b32 (truncating round).
// sel 0x07060302: byte0<-lo[2], byte1<-lo[3], byte2<-hi[2], byte3<-hi[3]
__device__ __forceinline__ unsigned pack2(float lo, float hi){
  return __builtin_amdgcn_perm(__float_as_uint(hi), __float_as_uint(lo), 0x07060302u);
}
__device__ __forceinline__ void top3_ins(float x, float& a, float& b, float& c){
  if (x > a){ c = b; b = a; a = x; }
  else if (x > b){ c = b; b = x; }
  else if (x > c){ c = x; }
}
__device__ __forceinline__ void load_frag(const __bf16* base, Frag& f){
  f.q[0] = *(const uint4*)base;
  f.q[1] = *(const uint4*)(base + 16);
}
__device__ __forceinline__ void atomic_max_f32(float* addr, float v){
  asm volatile("global_atomic_max_num_f32 %0, %1, off" :: "v"(addr), "v"(v) : "memory");
}

// ---------------------------------------------------------------------------
// k1: W[b,c,m] = sum_d ctx[b,c,d]*main[b,m,d]   (bf16 WMMA, fp32 accumulate)
// ---------------------------------------------------------------------------
__global__ __launch_bounds__(256) void k1_gemm_weights(
    const float* __restrict__ ctx, const float* __restrict__ mn,
    float* __restrict__ W){
  const int b  = blockIdx.z;
  const int c0 = blockIdx.x * 128;
  const int m0 = blockIdx.y * 128;
  __shared__ __align__(16) __bf16 lA[128*40];
  __shared__ __align__(16) __bf16 lB[128*40];
  const int t = threadIdx.x, lane = t & 31, w = t >> 5;
  const int wm = w & 3, wn = w >> 2;
  const int rowf = t >> 1;            // 0..127
  const int colg = (t & 1) * 16;      // 0 or 16
  const float* aSrc = ctx + ((size_t)b * LSEQ + (c0 + rowf)) * DDIM + colg;
  const float* bSrc = mn  + ((size_t)b * LSEQ + (m0 + rowf)) * DDIM + colg;

  v8f zero8 = {0,0,0,0,0,0,0,0};
  v8f acc[2][4];
  #pragma unroll
  for (int i = 0; i < 2; i++)
    #pragma unroll
    for (int j = 0; j < 4; j++) acc[i][j] = zero8;

  for (int kk = 0; kk < DDIM; kk += 32){
    __syncthreads();
    const float4* a4 = (const float4*)(aSrc + kk);
    const float4* b4 = (const float4*)(bSrc + kk);
    unsigned* la = (unsigned*)&lA[rowf*40 + colg];
    unsigned* lb = (unsigned*)&lB[rowf*40 + colg];
    #pragma unroll
    for (int q = 0; q < 4; q++){
      float4 av = a4[q]; float4 bv = b4[q];
      la[q*2+0] = pack2(av.x, av.y); la[q*2+1] = pack2(av.z, av.w);
      lb[q*2+0] = pack2(bv.x, bv.y); lb[q*2+1] = pack2(bv.z, bv.w);
    }
    __syncthreads();
    const int rA = lane & 15, kh = lane >> 4;
    Frag fa[2], fb[4];
    #pragma unroll
    for (int i = 0; i < 2; i++) load_frag(&lA[(wm*32 + i*16 + rA)*40 + kh*8], fa[i]);
    #pragma unroll
    for (int j = 0; j < 4; j++) load_frag(&lB[(wn*64 + j*16 + rA)*40 + kh*8], fb[j]);
    #pragma unroll
    for (int i = 0; i < 2; i++)
      #pragma unroll
      for (int j = 0; j < 4; j++)
        acc[i][j] = __builtin_amdgcn_wmma_f32_16x16x32_bf16(
            false, fa[i].v, false, fb[j].v, (short)0, acc[i][j], false, false);
  }
  const int nN = lane & 15, mh = (lane >> 4) * 8;
  #pragma unroll
  for (int i = 0; i < 2; i++)
    #pragma unroll
    for (int j = 0; j < 4; j++){
      float* wp = W + ((size_t)b*LSEQ + (c0 + wm*32 + i*16 + mh)) * LSEQ
                    + (m0 + wn*64 + j*16 + nN);
      #pragma unroll
      for (int r = 0; r < 8; r++) wp[(size_t)r * LSEQ] = acc[i][j][r];
    }
}

// ---------------------------------------------------------------------------
// k2: per-row (over m) stats: kth (3rd largest), 1/(v1+v2+v3), std
// ---------------------------------------------------------------------------
__global__ __launch_bounds__(256) void k2_row_stats(const float* __restrict__ W,
    float* __restrict__ kth, float* __restrict__ inv, float* __restrict__ stdo){
  const size_t row = blockIdx.x;                 // b*L + c
  const float4* r4 = (const float4*)(W + row * (size_t)LSEQ);
  const int t = threadIdx.x, lane = t & 31, w = t >> 5;
  float4 f0 = r4[t], f1 = r4[t + 256];
  float a = NEG_HUGE, b = NEG_HUGE, c = NEG_HUGE, s = 0.f, s2 = 0.f;
  float xs[8] = {f0.x,f0.y,f0.z,f0.w,f1.x,f1.y,f1.z,f1.w};
  #pragma unroll
  for (int e = 0; e < 8; e++){ float x = xs[e]; s += x; s2 += x*x; top3_ins(x,a,b,c); }
  #pragma unroll
  for (int off = 16; off >= 1; off >>= 1){
    float oa = __shfl_xor(a, off, 32), ob = __shfl_xor(b, off, 32), oc = __shfl_xor(c, off, 32);
    float os = __shfl_xor(s, off, 32), os2 = __shfl_xor(s2, off, 32);
    s += os; s2 += os2;
    top3_ins(oa,a,b,c); top3_ins(ob,a,b,c); top3_ins(oc,a,b,c);
  }
  __shared__ float red[8][5];
  if (lane == 0){ red[w][0]=a; red[w][1]=b; red[w][2]=c; red[w][3]=s; red[w][4]=s2; }
  __syncthreads();
  if (t == 0){
    float A=NEG_HUGE,B=NEG_HUGE,C=NEG_HUGE,S=0.f,S2=0.f;
    for (int q = 0; q < 8; q++){
      top3_ins(red[q][0],A,B,C); top3_ins(red[q][1],A,B,C); top3_ins(red[q][2],A,B,C);
      S += red[q][3]; S2 += red[q][4];
    }
    kth[row] = C;
    inv[row] = 1.f / (A + B + C);
    float mean = S * (1.f/LSEQ);
    float var  = S2 * (1.f/LSEQ) - mean*mean;
    stdo[row] = sqrtf(fmaxf(var, 0.f));
  }
}

// ---------------------------------------------------------------------------
// k3: per-column (over c) stats (for weights^T rows)
// ---------------------------------------------------------------------------
__global__ __launch_bounds__(256) void k3_col_stats(const float* __restrict__ W,
    float* __restrict__ kth, float* __restrict__ inv, float* __restrict__ stdo){
  const int b  = blockIdx.y;
  const int m0 = blockIdx.x * 64;
  const int t = threadIdx.x, mc = t & 63, strip = t >> 6;
  const float* base = W + (size_t)b * LSEQ * LSEQ + m0 + mc;
  float a=NEG_HUGE, bb=NEG_HUGE, c=NEG_HUGE, s=0.f, s2=0.f;
  for (int cc = strip; cc < LSEQ; cc += 4){
    float x = base[(size_t)cc * LSEQ];
    s += x; s2 += x*x; top3_ins(x,a,bb,c);
  }
  __shared__ float red[4][64][5];
  red[strip][mc][0]=a; red[strip][mc][1]=bb; red[strip][mc][2]=c;
  red[strip][mc][3]=s; red[strip][mc][4]=s2;
  __syncthreads();
  if (strip == 0){
    for (int q = 1; q < 4; q++){
      top3_ins(red[q][mc][0],a,bb,c); top3_ins(red[q][mc][1],a,bb,c); top3_ins(red[q][mc][2],a,bb,c);
      s += red[q][mc][3]; s2 += red[q][mc][4];
    }
    size_t gi = (size_t)b * LSEQ + m0 + mc;
    kth[gi] = c; inv[gi] = 1.f/(a + bb + c);
    float mean = s * (1.f/LSEQ);
    float var  = s2 * (1.f/LSEQ) - mean*mean;
    stdo[gi] = sqrtf(fmaxf(var, 0.f));
  }
}

// ---------------------------------------------------------------------------
// k4: masked-renormalized GEMM + fused std*|ref - att| epilogue
//   mode 0: out_m = std_m * |main - (maskC∘W)^T @ ctx|
//   mode 1: out_c = std_c * |ctx  - (maskM∘W)^T @ main|
// ---------------------------------------------------------------------------
__global__ __launch_bounds__(256) void k4_masked_gemm(
    const float* __restrict__ W,
    const float* __restrict__ ctx, const float* __restrict__ mn,
    const float* __restrict__ kth_c, const float* __restrict__ inv_c,
    const float* __restrict__ std_m,
    const float* __restrict__ kth_m, const float* __restrict__ inv_m,
    const float* __restrict__ std_c,
    float* __restrict__ out_m, float* __restrict__ out_c){
  const int b    = blockIdx.z >> 1;
  const int mode = blockIdx.z & 1;
  const int m0 = blockIdx.x * 128;
  const int d0 = blockIdx.y * 128;
  const float* Bsrc = mode ? mn  : ctx;
  const float* Ref  = mode ? ctx : mn;
  __shared__ __align__(16) __bf16 lA[128*40];
  __shared__ __align__(16) __bf16 lB[128*40];
  __shared__ float sKth[128], sInv[128], sStd[128];
  const int t = threadIdx.x, lane = t & 31, w = t >> 5;
  const int wm = w & 3, wn = w >> 2;
  const int p = t >> 4, mg = t & 15;
  if (t < 128){
    size_t gi = (size_t)b * LSEQ + m0 + t;
    sStd[t] = mode ? std_c[gi] : std_m[gi];
    sKth[t] = mode ? kth_m[gi] : 0.f;
    sInv[t] = mode ? inv_m[gi] : 0.f;
  }
  v8f zero8 = {0,0,0,0,0,0,0,0};
  v8f acc[2][4];
  #pragma unroll
  for (int i = 0; i < 2; i++)
    #pragma unroll
    for (int j = 0; j < 4; j++) acc[i][j] = zero8;

  for (int ck = 0; ck < LSEQ; ck += 32){
    __syncthreads();
    {   // A tile: masked/normalized W^T block, transposed into [m][c] layout
      const size_t rbase = ((size_t)b * LSEQ + ck + 2*p) * LSEQ + m0 + mg*8;
      float x0[8], x1[8];
      *(float4*)&x0[0] = *(const float4*)(W + rbase);
      *(float4*)&x0[4] = *(const float4*)(W + rbase + 4);
      *(float4*)&x1[0] = *(const float4*)(W + rbase + LSEQ);
      *(float4*)&x1[4] = *(const float4*)(W + rbase + LSEQ + 4);
      float k0 = 0.f, k1 = 0.f, i0 = 0.f, i1 = 0.f;
      if (mode == 0){
        size_t ci = (size_t)b * LSEQ + ck + 2*p;
        k0 = kth_c[ci];     i0 = inv_c[ci];
        k1 = kth_c[ci + 1]; i1 = inv_c[ci + 1];
      }
      #pragma unroll
      for (int e = 0; e < 8; e++){
        int ml = mg*8 + e;
        float v0, v1;
        if (mode == 0){
          v0 = (x0[e] >= k0) ? x0[e]*i0 : 0.f;
          v1 = (x1[e] >= k1) ? x1[e]*i1 : 0.f;
        } else {
          float km = sKth[ml], im = sInv[ml];
          v0 = (x0[e] >= km) ? x0[e]*im : 0.f;
          v1 = (x1[e] >= km) ? x1[e]*im : 0.f;
        }
        *(unsigned*)&lA[ml*40 + 2*p] = pack2(v0, v1);
      }
    }
    {   // B tile: ctx/main block transposed into [d][c] layout
      const size_t rbase = ((size_t)b * LSEQ + ck + 2*p) * DDIM + d0 + mg*8;
      float y0[8], y1[8];
      *(float4*)&y0[0] = *(const float4*)(Bsrc + rbase);
      *(float4*)&y0[4] = *(const float4*)(Bsrc + rbase + 4);
      *(float4*)&y1[0] = *(const float4*)(Bsrc + rbase + DDIM);
      *(float4*)&y1[4] = *(const float4*)(Bsrc + rbase + DDIM + 4);
      #pragma unroll
      for (int e = 0; e < 8; e++)
        *(unsigned*)&lB[(mg*8 + e)*40 + 2*p] = pack2(y0[e], y1[e]);
    }
    __syncthreads();
    const int rA = lane & 15, kh = lane >> 4;
    Frag fa[2], fb[4];
    #pragma unroll
    for (int i = 0; i < 2; i++) load_frag(&lA[(wm*32 + i*16 + rA)*40 + kh*8], fa[i]);
    #pragma unroll
    for (int j = 0; j < 4; j++) load_frag(&lB[(wn*64 + j*16 + rA)*40 + kh*8], fb[j]);
    #pragma unroll
    for (int i = 0; i < 2; i++)
      #pragma unroll
      for (int j = 0; j < 4; j++)
        acc[i][j] = __builtin_amdgcn_wmma_f32_16x16x32_bf16(
            false, fa[i].v, false, fb[j].v, (short)0, acc[i][j], false, false);
  }
  float* Out = mode ? out_c : out_m;
  const int nN = lane & 15, mh = (lane >> 4) * 8;
  #pragma unroll
  for (int i = 0; i < 2; i++)
    #pragma unroll
    for (int j = 0; j < 4; j++){
      const int mloc = wm*32 + i*16 + mh;
      const int d = d0 + wn*64 + j*16 + nN;
      #pragma unroll
      for (int r = 0; r < 8; r++){
        const int m = m0 + mloc + r;
        size_t gi = ((size_t)b * LSEQ + m) * DDIM + d;
        Out[gi] = sStd[mloc + r] * fabsf(Ref[gi] - acc[i][j][r]);
      }
    }
}

// ---------------------------------------------------------------------------
// k0: zero the output (atomic-max target)
// ---------------------------------------------------------------------------
__global__ void k0_zero(float* __restrict__ o, int n){
  int i = blockIdx.x * 256 + threadIdx.x;
  if (i < n) o[i] = 0.f;
}

// ---------------------------------------------------------------------------
// k5: conv1d(ks=3) as WMMA GEMM + bias + relu + global max pool (atomic max)
// ---------------------------------------------------------------------------
__global__ __launch_bounds__(256) void k5_conv(
    const float* __restrict__ Xc, const float* __restrict__ Xm,
    const float* __restrict__ cw, const float* __restrict__ cb,
    float* __restrict__ out){
  const int b  = blockIdx.z;
  const int ts = blockIdx.y;            // 0 -> outputs_c, 1 -> outputs_m
  const int l0 = blockIdx.x * 64;
  const float* X = ts ? Xm : Xc;
  __shared__ __align__(16) __bf16 lX[66*264];
  __shared__ __align__(16) __bf16 lW[128*40];
  __shared__ float sB[128];
  const int t = threadIdx.x, lane = t & 31, w = t >> 5;
  const int wm = w & 3, wn = w >> 2;
  const int p = t >> 4, fg = t & 15;
  if (t < 128) sB[t] = (t < NFILT) ? cb[t] : 0.f;
  for (int idx = t; idx < 66*64; idx += 256){
    int row = idx >> 6, c4 = idx & 63;
    int gl = l0 + row; if (gl > LSEQ-1) gl = LSEQ-1;
    float4 v = *(const float4*)(X + ((size_t)b*LSEQ + gl)*DDIM + c4*4);
    unsigned* dst = (unsigned*)&lX[row*264 + c4*4];
    dst[0] = pack2(v.x, v.y); dst[1] = pack2(v.z, v.w);
  }
  v8f zero8 = {0,0,0,0,0,0,0,0};
  v8f acc[4];
  #pragma unroll
  for (int j = 0; j < 4; j++) acc[j] = zero8;

  for (int step = 0; step < 24; step++){
    const int kappa = step * 32;
    const int kk  = kappa >> 8;         // kernel tap 0..2
    const int dd0 = kappa & 255;        // d offset within tap
    __syncthreads();
    {   // weight tile: wflat^T [f][32] from conv_w[k][d][f]
      const int dA = dd0 + 2*p, f0 = fg*8;
      const float* w0 = cw + ((size_t)kk*DDIM + dA) * NFILT + f0;
      const float* w1 = w0 + NFILT;
      #pragma unroll
      for (int e = 0; e < 8; e++){
        int f = f0 + e;
        float a0 = (f < NFILT) ? w0[e] : 0.f;
        float a1 = (f < NFILT) ? w1[e] : 0.f;
        *(unsigned*)&lW[f*40 + 2*p] = pack2(a0, a1);
      }
    }
    __syncthreads();
    const int rA = lane & 15, kh = lane >> 4;
    Frag fa, fb[4];
    load_frag(&lX[(wm*16 + rA + kk)*264 + dd0 + kh*8], fa);
    #pragma unroll
    for (int j = 0; j < 4; j++) load_frag(&lW[(wn*64 + j*16 + rA)*40 + kh*8], fb[j]);
    #pragma unroll
    for (int j = 0; j < 4; j++)
      acc[j] = __builtin_amdgcn_wmma_f32_16x16x32_bf16(
          false, fa.v, false, fb[j].v, (short)0, acc[j], false, false);
  }
  float* dst = out + b*(2*NFILT) + ts*NFILT;
  #pragma unroll
  for (int j = 0; j < 4; j++){
    const int f = wn*64 + j*16 + (lane & 15);
    float mx = 0.f;
    #pragma unroll
    for (int r = 0; r < 8; r++){
      const int l = l0 + wm*16 + r + 8*(lane >> 4);
      float v = acc[j][r] + sB[f];
      v = fmaxf(v, 0.f);
      if (l > LSEQ - KSZ) v = 0.f;     // conv valid: l <= 2045
      mx = fmaxf(mx, v);
    }
    mx = fmaxf(mx, __shfl_xor(mx, 16, 32));
    if ((lane < 16) && (f < NFILT)) atomic_max_f32(dst + f, mx);
  }
}

// ---------------------------------------------------------------------------
extern "C" void kernel_launch(void* const* d_in, const int* in_sizes, int n_in,
                              void* d_out, int out_size, void* d_ws, size_t ws_size,
                              hipStream_t stream){
  (void)in_sizes; (void)n_in; (void)out_size; (void)ws_size;
  const float* ctx = (const float*)d_in[0];
  const float* mn  = (const float*)d_in[1];
  const float* cw  = (const float*)d_in[2];
  const float* cb  = (const float*)d_in[3];
  float* out = (float*)d_out;

  // workspace layout (floats): W | 6 stat arrays | out_c | out_m  (~336 MB)
  float* W = (float*)d_ws;
  const size_t nBL = (size_t)NBATCH * LSEQ;
  float* kth_c = W + (size_t)NBATCH * LSEQ * LSEQ;
  float* inv_c = kth_c + nBL;
  float* std_c = inv_c + nBL;
  float* kth_m = std_c + nBL;
  float* inv_m = kth_m + nBL;
  float* std_m = inv_m + nBL;
  float* out_c = std_m + nBL;
  float* out_m = out_c + nBL * DDIM;

  k1_gemm_weights<<<dim3(LSEQ/128, LSEQ/128, NBATCH), 256, 0, stream>>>(ctx, mn, W);
  k2_row_stats   <<<dim3((unsigned)nBL), 256, 0, stream>>>(W, kth_c, inv_c, std_c);
  k3_col_stats   <<<dim3(LSEQ/64, NBATCH), 256, 0, stream>>>(W, kth_m, inv_m, std_m);
  k4_masked_gemm <<<dim3(LSEQ/128, DDIM/128, NBATCH*2), 256, 0, stream>>>(
      W, ctx, mn, kth_c, inv_c, std_m, kth_m, inv_m, std_c, out_m, out_c);
  k0_zero        <<<(NBATCH*2*NFILT + 255)/256, 256, 0, stream>>>(out, NBATCH*2*NFILT);
  k5_conv        <<<dim3(LSEQ/64, 2, NBATCH), 256, 0, stream>>>(out_c, out_m, cw, cb, out);
}
